// XLSTMMemoryCell_51719996179088
// MI455X (gfx1250) — matrix operand order repaired and checked by reading, not compiled, not run on hardware
//
#include <hip/hip_runtime.h>
#include <hip/hip_bf16.h>

typedef __attribute__((ext_vector_type(16))) __bf16 v16bf;
typedef __attribute__((ext_vector_type(8)))  float  v8f;

constexpr int kB = 64, kT = 512, kD = 256, kU = 512, kM = 64;
constexpr int kNC = 5 * kU + kM;   // 2624 fused output columns: i|f|o|c~|m|e
constexpr int kK  = kD + kU;       // 768 fused K: x-part then h-part
constexpr int kHS = 536;           // h LDS row stride (bf16), 16B aligned, bank-spread
constexpr int kXS = 264;           // x LDS row stride (bf16), 16B aligned, bank-spread

struct alignas(16) U128 { unsigned int w[4]; };
union FragAB { v16bf v; U128 q[2]; };

__device__ __forceinline__ float clipf(float v)    { return fminf(10.f, fmaxf(-10.f, v)); }
__device__ __forceinline__ float sigmoidf(float v) { return 1.f / (1.f + __expf(-v)); }

// Pack [Wi|Wf|Wo|Wc|Wm|We ; Ui|Uf|Uo|Uc|0|Ue] transposed to WK[col][k] bf16.
__global__ void prep_weights(const float* Wi, const float* Ui, const float* Wf, const float* Uf,
                             const float* Wo, const float* Uo, const float* Wc, const float* Uc,
                             const float* Wm, const float* We, const float* Ue, __bf16* WK) {
  int idx = blockIdx.x * blockDim.x + threadIdx.x;
  if (idx >= kNC * kK) return;
  int col = idx / kK, kk = idx % kK;
  float v;
  if (col < 5 * kU) {
    int seg = col >> 9, u = col & (kU - 1);
    if (kk < kD) {
      const float* W = seg == 0 ? Wi : seg == 1 ? Wf : seg == 2 ? Wo : seg == 3 ? Wc : Wm;
      v = W[kk * kU + u];
    } else {
      int kh = kk - kD;
      if (seg < 4) {
        const float* U = seg == 0 ? Ui : seg == 1 ? Uf : seg == 2 ? Uo : Uc;
        v = U[kh * kU + u];
      } else v = 0.f;                       // Wm has no recurrent term
    }
  } else {
    int m = col - 5 * kU;
    v = (kk < kD) ? We[kk * kM + m] : Ue[(kk - kD) * kM + m];
  }
  WK[(size_t)col * kK + kk] = (__bf16)v;
}

__global__ void prep_x(const float* x, __bf16* Xc, int n) {
  int i = blockIdx.x * blockDim.x + threadIdx.x;
  if (i < n) Xc[i] = (__bf16)clipf(x[i]);
}

__launch_bounds__(512)
__global__ void xlstm_scan(const __bf16* __restrict__ Xc, const __bf16* __restrict__ WK,
                           const float* __restrict__ bi, const float* __restrict__ bfv,
                           const float* __restrict__ bo, const float* __restrict__ bc,
                           const float* __restrict__ be,
                           float* __restrict__ mem,   // [4][16][64][512] ring buffers
                           float* __restrict__ out) { // [B][T][U]
  __shared__ __bf16 h_bf[16 * kHS];
  __shared__ __bf16 x_lds[2][16 * kXS];
  __shared__ float  e_buf[16][kM];

  const int tid = threadIdx.x, wv = tid >> 5, lane = tid & 31;
  const int lhi = lane >> 4, llo = lane & 15;
  const int b0 = blockIdx.x * 16;

  for (int i = tid; i < 16 * kHS; i += blockDim.x) h_bf[i] = (__bf16)0.f;

  const int u0 = wv * 16 + llo, u1 = (wv + 16) * 16 + llo;   // owned column blocks
  const int ecol = 5 * kU + (wv & 3) * 16 + llo;             // e-tile column (waves 0..3)
  const bool do_e = (wv < 4);
  const float bi0 = bi[u0],  bi1 = bi[u1];
  const float bf0 = bfv[u0], bf1 = bfv[u1];
  const float bo0 = bo[u0],  bo1 = bo[u1];
  const float bc0 = bc[u0],  bc1 = bc[u1];
  const float bee = do_e ? be[(wv & 3) * 16 + llo] : 0.f;

  float* memBlk = mem + (size_t)blockIdx.x * 16 * kM * kU;
  const v8f zero8 = {0.f,0.f,0.f,0.f,0.f,0.f,0.f,0.f};

  // Async stage of one 16x256 bf16 x-tile (8KB) into LDS: one b128 per thread.
  auto stage_x = [&](int bb, int t) {
    int r = tid >> 5, c = tid & 31;
    const __bf16* src = Xc + ((size_t)(b0 + r) * kT + t) * kD + c * 8;
    unsigned dst = (unsigned)(size_t)&x_lds[bb][r * kXS + c * 8];
    asm volatile("global_load_async_to_lds_b128 %0, %1, off"
                 :: "v"(dst), "v"((unsigned long long)src) : "memory");
  };
  auto wait_async0 = [&] { asm volatile("s_wait_asynccnt 0x0" ::: "memory"); };

  // One GEMM phase: [16x768] @ [768 x 5(+1)*16] bf16 WMMA with 2-slot B pipeline.
  auto gemm_phase = [&](const __bf16* wk, const __bf16* xb, int ucol,
                        bool want_e, v8f (&acc)[6]) {
    const __bf16* wb0 = wk + (size_t)(0 * kU + ucol) * kK;
    const __bf16* wb1 = wk + (size_t)(1 * kU + ucol) * kK;
    const __bf16* wb2 = wk + (size_t)(2 * kU + ucol) * kK;
    const __bf16* wb3 = wk + (size_t)(3 * kU + ucol) * kK;
    const __bf16* wb4 = wk + (size_t)(4 * kU + ucol) * kK;
    const __bf16* wbe = wk + (size_t)ecol * kK;
    FragAB Bf[2][6];

#define LOADB(SLOT, KS) do {                                                  \
    const int kb_ = (KS) * 32 + lhi * 16;                                     \
    Bf[SLOT][0].q[0] = *reinterpret_cast<const U128*>(wb0 + kb_);             \
    Bf[SLOT][0].q[1] = *reinterpret_cast<const U128*>(wb0 + kb_ + 8);         \
    Bf[SLOT][1].q[0] = *reinterpret_cast<const U128*>(wb1 + kb_);             \
    Bf[SLOT][1].q[1] = *reinterpret_cast<const U128*>(wb1 + kb_ + 8);         \
    Bf[SLOT][2].q[0] = *reinterpret_cast<const U128*>(wb2 + kb_);             \
    Bf[SLOT][2].q[1] = *reinterpret_cast<const U128*>(wb2 + kb_ + 8);         \
    Bf[SLOT][3].q[0] = *reinterpret_cast<const U128*>(wb3 + kb_);             \
    Bf[SLOT][3].q[1] = *reinterpret_cast<const U128*>(wb3 + kb_ + 8);         \
    Bf[SLOT][4].q[0] = *reinterpret_cast<const U128*>(wb4 + kb_);             \
    Bf[SLOT][4].q[1] = *reinterpret_cast<const U128*>(wb4 + kb_ + 8);         \
    if (want_e) {                                                             \
      Bf[SLOT][5].q[0] = *reinterpret_cast<const U128*>(wbe + kb_);           \
      Bf[SLOT][5].q[1] = *reinterpret_cast<const U128*>(wbe + kb_ + 8);       \
    } } while (0)

#define DOWMMA(SLOT, KS) do {                                                 \
    const int ks_ = (KS);                                                     \
    FragAB A_;                                                                \
    const __bf16* ap_ = (ks_ < 8)                                             \
        ? (xb + llo * kXS + ks_ * 32 + lhi * 16)                              \
        : (&h_bf[llo * kHS + (ks_ - 8) * 32 + lhi * 16]);                     \
    A_.q[0] = *reinterpret_cast<const U128*>(ap_);                            \
    A_.q[1] = *reinterpret_cast<const U128*>(ap_ + 8);                        \
    acc[0] = __builtin_amdgcn_wmma_f32_16x16x32_bf16(false, A_.v, false, Bf[SLOT][0].v, (short)0, acc[0], false, false); \
    acc[1] = __builtin_amdgcn_wmma_f32_16x16x32_bf16(false, A_.v, false, Bf[SLOT][1].v, (short)0, acc[1], false, false); \
    acc[2] = __builtin_amdgcn_wmma_f32_16x16x32_bf16(false, A_.v, false, Bf[SLOT][2].v, (short)0, acc[2], false, false); \
    acc[3] = __builtin_amdgcn_wmma_f32_16x16x32_bf16(false, A_.v, false, Bf[SLOT][3].v, (short)0, acc[3], false, false); \
    acc[4] = __builtin_amdgcn_wmma_f32_16x16x32_bf16(false, A_.v, false, Bf[SLOT][4].v, (short)0, acc[4], false, false); \
    if (want_e)                                                               \
      acc[5] = __builtin_amdgcn_wmma_f32_16x16x32_bf16(false, A_.v, false, Bf[SLOT][5].v, (short)0, acc[5], false, false); \
    } while (0)

    LOADB(0, 0);
    #pragma unroll 1
    for (int kp = 0; kp < 11; ++kp) {
      LOADB(1, 2 * kp + 1);
      DOWMMA(0, 2 * kp);
      LOADB(0, 2 * kp + 2);
      DOWMMA(1, 2 * kp + 1);
    }
    LOADB(1, 23);
    DOWMMA(0, 22);
    DOWMMA(1, 23);
#undef LOADB
#undef DOWMMA
  };

  int head = 0;
  stage_x(0, 0);
  wait_async0();
  __syncthreads();

  #pragma unroll 1
  for (int t = 0; t < kT; ++t) {
    // opaque zero offset: blocks hoisting of t-invariant WK loads, keeps global AS
    size_t lz = 0;
    asm volatile("" : "+s"(lz));
    const __bf16* wk = WK + lz;

    if (t + 1 < kT) stage_x((t + 1) & 1, t + 1);       // overlap next x with this step
    const __bf16* xb = x_lds[t & 1];

    // ---- phase 1: GEMM for cb0 (+ e tiles on waves 0..3) ----
    v8f acc[6];
    #pragma unroll
    for (int i = 0; i < 6; ++i) acc[i] = zero8;
    gemm_phase(wk, xb, u0, do_e, acc);

    if (do_e) {
      #pragma unroll
      for (int j = 0; j < 8; ++j) e_buf[j + 8 * lhi][(wv & 3) * 16 + llo] = acc[5][j] + bee;
    }
    __syncthreads();

    if (tid < 16) {                                    // softmax(clip(e)) per row
      int r = tid; float mx = -3.4e38f;
      for (int m = 0; m < kM; ++m) { float v = clipf(e_buf[r][m]); e_buf[r][m] = v; mx = fmaxf(mx, v); }
      float s = 0.f;
      for (int m = 0; m < kM; ++m) { float v = __expf(e_buf[r][m] - mx); e_buf[r][m] = v; s += v; }
      float inv = 1.f / s;
      for (int m = 0; m < kM; ++m) e_buf[r][m] *= inv;
    }
    __syncthreads();

    const int headN = (head + 63) & 63;
    float h0s[8], h1s[8];

    // ---- phase 2: attention + gates + state update for cb0 ----
    #pragma unroll
    for (int j = 0; j < 8; ++j) {
      const int r = j + 8 * lhi;
      float* mrow = memBlk + (size_t)r * (kM * kU);
      float att = 0.f;
      for (int m = 0; m < kM; ++m)
        att = fmaf(mrow[((head + m) & 63) * kU + u0], e_buf[r][m], att);
      float gi = sigmoidf(clipf(acc[0][j] + bi0)), gf = sigmoidf(clipf(acc[1][j] + bf0));
      float go = sigmoidf(clipf(acc[2][j] + bo0)), ct = tanhf(clipf(acc[3][j] + bc0));
      float c = gf * (att + acc[4][j]) + gi * ct;
      mrow[headN * kU + u0] = c;
      float hn = go * tanhf(c);
      out[((size_t)(b0 + r) * kT + t) * kU + u0] = hn;
      h0s[j] = hn;
    }

    // ---- phase 3: GEMM for cb1 (reuses acc registers) ----
    #pragma unroll
    for (int i = 0; i < 5; ++i) acc[i] = zero8;
    gemm_phase(wk, xb, u1, false, acc);

    // ---- phase 4: attention + gates + state update for cb1 ----
    #pragma unroll
    for (int j = 0; j < 8; ++j) {
      const int r = j + 8 * lhi;
      float* mrow = memBlk + (size_t)r * (kM * kU);
      float att = 0.f;
      for (int m = 0; m < kM; ++m)
        att = fmaf(mrow[((head + m) & 63) * kU + u1], e_buf[r][m], att);
      float gi = sigmoidf(clipf(acc[0][j] + bi1)), gf = sigmoidf(clipf(acc[1][j] + bf1));
      float go = sigmoidf(clipf(acc[2][j] + bo1)), ct = tanhf(clipf(acc[3][j] + bc1));
      float c = gf * (att + acc[4][j]) + gi * ct;
      mrow[headN * kU + u1] = c;
      float hn = go * tanhf(c);
      out[((size_t)(b0 + r) * kT + t) * kU + u1] = hn;
      h1s[j] = hn;
    }

    __syncthreads();                                   // all h_bf readers (phase 3) done
    #pragma unroll
    for (int j = 0; j < 8; ++j) {
      const int r = j + 8 * lhi;
      h_bf[r * kHS + u0] = (__bf16)h0s[j];
      h_bf[r * kHS + u1] = (__bf16)h1s[j];
    }
    head = headN;
    wait_async0();                                     // x(t+1) tile landed in LDS
    __syncthreads();
  }
}

extern "C" void kernel_launch(void* const* d_in, const int* in_sizes, int n_in,
                              void* d_out, int out_size, void* d_ws, size_t ws_size,
                              hipStream_t stream) {
  const float* x  = (const float*)d_in[0];
  const float* Wi = (const float*)d_in[1];
  const float* Ui = (const float*)d_in[2];
  const float* bi = (const float*)d_in[3];
  const float* Wf = (const float*)d_in[4];
  const float* Uf = (const float*)d_in[5];
  const float* bf = (const float*)d_in[6];
  const float* Wo = (const float*)d_in[7];
  const float* Uo = (const float*)d_in[8];
  const float* bo = (const float*)d_in[9];
  const float* Wc = (const float*)d_in[10];
  const float* Uc = (const float*)d_in[11];
  const float* bc = (const float*)d_in[12];
  const float* Wm = (const float*)d_in[13];
  const float* We = (const float*)d_in[14];
  const float* Ue = (const float*)d_in[15];
  const float* be = (const float*)d_in[16];

  char* ws = (char*)d_ws;
  __bf16* WK  = (__bf16*)ws;                                  // 2624*768*2  = 4,030,464 B
  __bf16* Xc  = (__bf16*)(ws + 4030464);                      // 64*512*256*2 = 16,777,216 B
  float*  mem = (float*)(ws + 4030464 + 16777216);            // 64*64*512*4  = 8,388,608 B

  (void)hipMemsetAsync(mem, 0, (size_t)kB * kM * kU * sizeof(float), stream);

  { int tot = kNC * kK;
    prep_weights<<<(tot + 255) / 256, 256, 0, stream>>>(Wi, Ui, Wf, Uf, Wo, Uo, Wc, Uc, Wm, We, Ue, WK); }
  { int n = kB * kT * kD;
    prep_x<<<(n + 255) / 256, 256, 0, stream>>>(x, Xc, n); }

  xlstm_scan<<<kB / 16, 512, 0, stream>>>(Xc, WK, bi, bf, bo, bc, be, mem, (float*)d_out);
}
